// SwinMSA_65730179498057
// MI455X (gfx1250) — matrix operand order repaired and checked by reading, not compiled, not run on hardware
//
#include <hip/hip_runtime.h>

#define EMBED_DIM 256
#define NUM_HEADS 8
#define HEAD_DIM 32
#define NTOK 49            // window tokens
#define NPAD 64            // padded tokens (4 tiles of 16)
#define BWIN 4096          // number of windows

typedef __attribute__((ext_vector_type(16))) __bf16 v16bf;
typedef __attribute__((ext_vector_type(8)))  float  v8f;

// ---------------- helpers ----------------

__device__ __forceinline__ unsigned short f2bf(float f) {
    unsigned int u = __float_as_uint(f);
    u = (u + 0x7FFFu + ((u >> 16) & 1u)) >> 16;   // round-to-nearest-even
    return (unsigned short)u;
}

// A-matrix fragment (16x32 bf16, M x K). Element (m,k) at p[m*stride + k].
// ISA layout: lanes 0-15: M=lane, V0..3 -> K=0..7 (pairs), V4..7 -> K=16..23
//             lanes 16-31: M=lane-16, V0..3 -> K=8..15, V4..7 -> K=24..31
__device__ __forceinline__ v16bf load_a_frag(const unsigned short* p, int stride, int lane) {
    int m = lane & 15, half = lane >> 4;
    const unsigned short* row = p + m * stride;
    union { unsigned int u[8]; v16bf v; } r;
#pragma unroll
    for (int i = 0; i < 8; ++i) {
        int kb = ((i < 4) ? 0 : 16) + half * 8 + (i & 3) * 2;
        r.u[i] = *(const unsigned int*)(row + kb);
    }
    return r.v;
}

// B-matrix fragment (32x16 bf16, K x N). Element (k,n) at p[n*stride + k].
// lanes 0-15: N=lane, V0..3 -> K=0..7, V4..7 -> K=16..23; lanes 16-31: N=lane-16, V0..3 -> K=8..15, V4..7 -> K=24..31.
__device__ __forceinline__ v16bf load_b_frag(const unsigned short* p, int stride, int lane) {
    int n = lane & 15;
    const unsigned int* col = (const unsigned int*)(p + n * stride);
    union { unsigned int u[8]; v16bf v; } r;
#pragma unroll
    for (int i = 0; i < 8; ++i) r.u[i] = col[((i < 4) ? 0 : 8) + (lane >> 4) * 4 + (i & 3)];
    return r.v;
}

__device__ __forceinline__ v8f wmma_bf16(v16bf a, v16bf b, v8f c) {
    return __builtin_amdgcn_wmma_f32_16x16x32_bf16(false, a, false, b, (short)0, c, false, false);
}

// ---------------- prep: weights -> bf16, fused bias table ----------------
// ws layout: [0)            wq_bf   768*256 bf16
//            [393216)       wp_bf   256*256 bf16
//            [524288)       biasf   8*64*64 fp32
__global__ void swin_prep_kernel(const float* __restrict__ wqkv,
                                 const float* __restrict__ wproj,
                                 const float* __restrict__ btab,
                                 const int*   __restrict__ bidx,
                                 unsigned short* __restrict__ wq_bf,
                                 unsigned short* __restrict__ wp_bf,
                                 float* __restrict__ biasf) {
    const int NW1 = 768 * 256;         // 196608
    const int NW2 = 256 * 256;         // 65536
    const int NB  = 8 * 64 * 64;       // 32768
    int i = blockIdx.x * blockDim.x + threadIdx.x;
    for (; i < NW1 + NW2 + NB; i += gridDim.x * blockDim.x) {
        if (i < NW1) {
            wq_bf[i] = f2bf(wqkv[i]);
        } else if (i < NW1 + NW2) {
            wp_bf[i - NW1] = f2bf(wproj[i - NW1]);
        } else {
            int j = i - NW1 - NW2;
            int h = j >> 12, q = (j >> 6) & 63, k = j & 63;
            float v = 0.f;
            if (q < NTOK && k < NTOK) v = btab[bidx[q * NTOK + k] * NUM_HEADS + h];
            biasf[j] = v;
        }
    }
}

// ---------------- fused window-MSA kernel: one workgroup per window -------
__global__ void __launch_bounds__(256)
swin_msa_kernel(const float* __restrict__ x,
                const float* __restrict__ b_qkv,
                const float* __restrict__ b_proj,
                const unsigned short* __restrict__ wq_bf,
                const unsigned short* __restrict__ wp_bf,
                const float* __restrict__ biasf,
                float* __restrict__ out) {
    // 128 KB LDS, regions aliased across stages:
    //  [0,      16384)  xb   (x in bf16, 64x256)   -> later ob (attn out 64x256)
    //  [16384,  32768)  qs   ([8][64][32])         -> later ps lower half
    //  [32768,  49152)  ks   ([8][64][32])         -> later ps upper half
    //  [49152,  65536)  vT   ([8][32][64])  persistent
    __shared__ unsigned short smem[65536];
    unsigned short* xb = smem;
    unsigned short* ob = smem;
    unsigned short* qs = smem + 16384;
    unsigned short* ks = smem + 32768;
    unsigned short* ps = smem + 16384;   // 8*64*64 bf16, aliases qs+ks
    unsigned short* vT = smem + 49152;

    const int b    = blockIdx.x;
    const int tid  = threadIdx.x;
    const int wv   = tid >> 5;
    const int lane = tid & 31;
    const int n    = lane & 15;          // D-frag column / B-frag column
    const int mb   = (lane >> 4) * 8;    // D-frag row base (lanes 16-31 -> rows 8..15)

    // ---- stage 0: load x window, fp32 -> bf16, pad rows 49..63 with 0 ----
    const float* xw = x + (size_t)b * NTOK * EMBED_DIM;
    for (int idx = tid; idx < NPAD * (EMBED_DIM / 4); idx += 256) {
        int row = idx >> 6;
        int c4  = (idx & 63) << 2;
        float4 f = make_float4(0.f, 0.f, 0.f, 0.f);
        if (row < NTOK) f = *(const float4*)(xw + row * EMBED_DIM + c4);
        unsigned short* d = xb + row * EMBED_DIM + c4;
        d[0] = f2bf(f.x); d[1] = f2bf(f.y); d[2] = f2bf(f.z); d[3] = f2bf(f.w);
    }
    __syncthreads();

    // ---- stage 1: QKV GEMM  [64x256] x [256 -> 768], scatter to q/k/vT ----
    // wave wv handles output-feature tiles nt = wv*6 .. wv*6+5 (48 tiles total)
    for (int t = 0; t < 6; ++t) {
        int nt = wv * 6 + t;
        int F0 = nt * 16;
        v8f acc[4];
#pragma unroll
        for (int it = 0; it < 4; ++it) acc[it] = (v8f){0,0,0,0,0,0,0,0};
        for (int ksp = 0; ksp < 8; ++ksp) {
            v16bf bfrag = load_b_frag(wq_bf + F0 * 256 + ksp * 32, 256, lane);
#pragma unroll
            for (int it = 0; it < 4; ++it) {
                v16bf afrag = load_a_frag(xb + it * 16 * 256 + ksp * 32, 256, lane);
                acc[it] = wmma_bf16(afrag, bfrag, acc[it]);
            }
        }
        // feature f = h*96 + sel*32 + c  (reshape B,N,H,3,C)
        int h   = F0 / 96;
        int sel = (F0 % 96) / 32;
        int c0  = F0 & 31;               // 0 or 16
        float bias = b_qkv[F0 + n];
#pragma unroll
        for (int it = 0; it < 4; ++it) {
#pragma unroll
            for (int v = 0; v < 8; ++v) {
                int tok = it * 16 + mb + v;
                unsigned short val = f2bf(acc[it][v] + bias);
                if (sel == 0)      qs[(h * 64 + tok) * 32 + c0 + n] = val;
                else if (sel == 1) ks[(h * 64 + tok) * 32 + c0 + n] = val;
                else               vT[(h * 32 + c0 + n) * 64 + tok] = val;
            }
        }
    }
    __syncthreads();

    // ---- stage 2a: energy + softmax, wave wv owns head h = wv -------------
    const int h = wv;
    v16bf aq[4], bk[4];
#pragma unroll
    for (int jt = 0; jt < 4; ++jt) {
        aq[jt] = load_a_frag(qs + (h * 64 + jt * 16) * 32, 32, lane);  // q rows
        bk[jt] = load_b_frag(ks + (h * 64 + jt * 16) * 32, 32, lane);  // k^T cols
    }
    __syncthreads();   // all fragments read before ps overwrites qs/ks

    const float inv32   = 0.03125f;           // scale/QUERY_VALUE = 1/32
    const float invs32  = 0.17677669529663687f; // 1/QUERY_VALUE = 1/sqrt(32)
#pragma unroll
    for (int it = 0; it < 4; ++it) {
        v8f e[4];
#pragma unroll
        for (int jt = 0; jt < 4; ++jt)
            e[jt] = wmma_bf16(aq[it], bk[jt], (v8f){0,0,0,0,0,0,0,0});

        float xv[4][8];
#pragma unroll
        for (int jt = 0; jt < 4; ++jt) {
#pragma unroll
            for (int v = 0; v < 8; ++v) {
                int col = jt * 16 + n;
                int row = it * 16 + mb + v;
                float bg = biasf[(h * 64 + row) * 64 + col];
                float val = e[jt][v] * inv32 + bg * invs32;
                xv[jt][v] = (col < NTOK) ? val : -3.0e38f;
            }
        }
#pragma unroll
        for (int v = 0; v < 8; ++v) {
            float rm = fmaxf(fmaxf(xv[0][v], xv[1][v]), fmaxf(xv[2][v], xv[3][v]));
            rm = fmaxf(rm, __shfl_xor(rm, 1));
            rm = fmaxf(rm, __shfl_xor(rm, 2));
            rm = fmaxf(rm, __shfl_xor(rm, 4));
            rm = fmaxf(rm, __shfl_xor(rm, 8));   // stays within 16-lane half (= one row)
            float s = 0.f;
#pragma unroll
            for (int jt = 0; jt < 4; ++jt) { xv[jt][v] = __expf(xv[jt][v] - rm); s += xv[jt][v]; }
            s += __shfl_xor(s, 1);
            s += __shfl_xor(s, 2);
            s += __shfl_xor(s, 4);
            s += __shfl_xor(s, 8);
            float inv = 1.0f / s;
            int row = it * 16 + mb + v;
#pragma unroll
            for (int jt = 0; jt < 4; ++jt)
                ps[(h * 64 + row) * 64 + jt * 16 + n] = f2bf(xv[jt][v] * inv);
        }
    }

    // ---- stage 2b: out = P @ V  (K over 64 tokens, 2 bf16 WMMA steps) ----
#pragma unroll
    for (int it = 0; it < 4; ++it) {
        v8f o0 = (v8f){0,0,0,0,0,0,0,0}, o1 = (v8f){0,0,0,0,0,0,0,0};
        for (int ksp = 0; ksp < 2; ++ksp) {
            v16bf ap  = load_a_frag(ps + (h * 64 + it * 16) * 64 + ksp * 32, 64, lane);
            v16bf bv0 = load_b_frag(vT + (h * 32 +  0) * 64 + ksp * 32, 64, lane);
            v16bf bv1 = load_b_frag(vT + (h * 32 + 16) * 64 + ksp * 32, 64, lane);
            o0 = wmma_bf16(ap, bv0, o0);
            o1 = wmma_bf16(ap, bv1, o1);
        }
#pragma unroll
        for (int v = 0; v < 8; ++v) {
            int tok = it * 16 + mb + v;
            ob[tok * 256 + h * 32 + n]      = f2bf(o0[v]);
            ob[tok * 256 + h * 32 + 16 + n] = f2bf(o1[v]);
        }
    }
    __syncthreads();

    // ---- stage 3: projection  [64x256] x [256x256]^T + b_proj ------------
    for (int t = 0; t < 2; ++t) {
        int nt = wv * 2 + t;
        int F0 = nt * 16;
        v8f acc[4];
#pragma unroll
        for (int it = 0; it < 4; ++it) acc[it] = (v8f){0,0,0,0,0,0,0,0};
        for (int ksp = 0; ksp < 8; ++ksp) {
            v16bf bfrag = load_b_frag(wp_bf + F0 * 256 + ksp * 32, 256, lane);
#pragma unroll
            for (int it = 0; it < 4; ++it) {
                v16bf afrag = load_a_frag(ob + it * 16 * 256 + ksp * 32, 256, lane);
                acc[it] = wmma_bf16(afrag, bfrag, acc[it]);
            }
        }
        float bias = b_proj[F0 + n];
#pragma unroll
        for (int it = 0; it < 4; ++it) {
#pragma unroll
            for (int v = 0; v < 8; ++v) {
                int tok = it * 16 + mb + v;
                if (tok < NTOK)
                    out[((size_t)b * NTOK + tok) * 256 + F0 + n] = acc[it][v] + bias;
            }
        }
    }
}

// ---------------- launch -------------------------------------------------
extern "C" void kernel_launch(void* const* d_in, const int* in_sizes, int n_in,
                              void* d_out, int out_size, void* d_ws, size_t ws_size,
                              hipStream_t stream) {
    (void)in_sizes; (void)n_in; (void)out_size; (void)ws_size;
    const float* x      = (const float*)d_in[0];
    const float* w_qkv  = (const float*)d_in[1];
    const float* b_qkv  = (const float*)d_in[2];
    const float* btab   = (const float*)d_in[3];
    const int*   bidx   = (const int*)d_in[4];
    const float* w_proj = (const float*)d_in[5];
    const float* b_proj = (const float*)d_in[6];
    float*       out    = (float*)d_out;

    unsigned short* wq_bf = (unsigned short*)d_ws;                 // 768*256 bf16
    unsigned short* wp_bf = wq_bf + 768 * 256;                     // 256*256 bf16
    float*          biasf = (float*)(wp_bf + 256 * 256);           // 8*64*64 fp32

    swin_prep_kernel<<<1152, 256, 0, stream>>>(w_qkv, w_proj, btab, bidx,
                                               wq_bf, wp_bf, biasf);
    swin_msa_kernel<<<BWIN, 256, 0, stream>>>(x, b_qkv, b_proj,
                                              wq_bf, wp_bf, biasf, out);
}
